// HGDCNet_17231408792164
// MI455X (gfx1250) — matrix operand order repaired and checked by reading, not compiled, not run on hardware
//
#include <hip/hip_runtime.h>
#include <hip/hip_bf16.h>

#define HH 100       // hidden width
#define NTILES 7     // 7 x 16 = 112 >= 100

typedef __attribute__((ext_vector_type(16))) __bf16 bf16x16;
typedef __attribute__((ext_vector_type(8)))  float  f32x8;

__device__ __forceinline__ unsigned short f32_to_bf16_bits(float x) {
  unsigned u = __builtin_bit_cast(unsigned, x);
  return (unsigned short)((u + 0x7FFFu + ((u >> 16) & 1u)) >> 16); // RNE
}
__device__ __forceinline__ __bf16 bits_to_bf16(unsigned short s) {
  return __builtin_bit_cast(__bf16, s);
}

// ---------------- per-destination degree count ----------------
__global__ void hgd_count_kernel(const int* __restrict__ col, int E, unsigned* __restrict__ cnt) {
  int e = blockIdx.x * blockDim.x + threadIdx.x;
  if (e < E) atomicAdd(&cnt[col[e]], 1u);
}

// deg^{-1/2} (0 if deg == 0)
__global__ void hgd_dis_kernel(const unsigned* __restrict__ cnt, float* __restrict__ dis, int n) {
  int i = blockIdx.x * blockDim.x + threadIdx.x;
  if (i < n) {
    unsigned d = cnt[i];
    dis[i] = (d > 0u) ? rsqrtf((float)d) : 0.0f;
  }
}

// ---------------- single-workgroup exclusive scan: offs[0..n], offs[n] = total ----------------
__global__ void __launch_bounds__(1024) hgd_scan_kernel(const unsigned* __restrict__ cnt,
                                                        int* __restrict__ offs, int n) {
  __shared__ int sbuf[1024];
  __shared__ int carry;
  const int tid = threadIdx.x;
  if (tid == 0) carry = 0;
  __syncthreads();
  for (int base = 0; base < n; base += 1024) {
    int x = (base + tid < n) ? (int)cnt[base + tid] : 0;
    sbuf[tid] = x;
    __syncthreads();
    #pragma unroll
    for (int o = 1; o < 1024; o <<= 1) {
      int v = (tid >= o) ? sbuf[tid - o] : 0;
      __syncthreads();
      sbuf[tid] += v;
      __syncthreads();
    }
    int incl = sbuf[tid];
    int run  = carry;
    if (base + tid < n) offs[base + tid] = run + incl - x;
    __syncthreads();
    if (tid == 1023) carry = run + incl;
    __syncthreads();
  }
  if (tid == 0) offs[n] = carry;
}

__global__ void hgd_copy_int(int* __restrict__ dst, const int* __restrict__ src, int n) {
  int i = blockIdx.x * blockDim.x + threadIdx.x;
  if (i < n) dst[i] = src[i];
}

// bucket fill: CSR by destination; also precompute permuted edge norm
__global__ void hgd_fill_kernel(const int* __restrict__ row, const int* __restrict__ col,
                                const float* __restrict__ dis,
                                int* __restrict__ cur, int* __restrict__ prow,
                                float* __restrict__ pnorm, int E) {
  int e = blockIdx.x * blockDim.x + threadIdx.x;
  if (e < E) {
    int r = row[e], c = col[e];
    int pos = atomicAdd(&cur[c], 1);
    prow[pos]  = r;
    pnorm[pos] = dis[r] * dis[c];
  }
}

// ---------------- bf16 WMMA GEMM: Y[m, yoff+0..99] = X[m,:K] @ W[K x 100] ----------------
// 256 threads = 8 waves; wave w -> rows [blk*128 + 16w, +16), cols 0..111 (7 wmma tiles)
__global__ void __launch_bounds__(256) hgd_gemm_wmma(
    const float* __restrict__ X, int ldx, int K,
    const float* __restrict__ W,
    float* __restrict__ Y, int ldy, int yoff, int nrows,
    const float* __restrict__ bias, int relu)
{
  __shared__ unsigned short As[128][34];  // pad 34: 17-dword row stride, conflict-free
  __shared__ unsigned short Bs[112][34];  // transposed: Bs[n][k] = W[kb+k][n]

  const int tid  = threadIdx.x;
  const int wave = tid >> 5;
  const int lane = tid & 31;
  const int half = lane >> 4;
  const int l16  = lane & 15;
  const long mBase = (long)blockIdx.x * 128;

  f32x8 acc[NTILES];
  #pragma unroll
  for (int t = 0; t < NTILES; t++) acc[t] = (f32x8)(0.0f);

  for (int kb = 0; kb < K; kb += 32) {
    for (int i = tid; i < 128 * 32; i += 256) {
      int r = i >> 5, c = i & 31;
      long m = mBase + r;
      int k = kb + c;
      float v = (m < nrows && k < K) ? X[m * (long)ldx + k] : 0.0f;
      As[r][c] = f32_to_bf16_bits(v);
    }
    for (int i = tid; i < 112 * 32; i += 256) {
      int nn = i >> 5, c = i & 31;
      int k = kb + c;
      float v = (nn < HH && k < K) ? W[(long)k * HH + nn] : 0.0f;
      Bs[nn][c] = f32_to_bf16_bits(v);
    }
    __syncthreads();

    // A fragment per ISA 16-bit 16x32 layout
    bf16x16 a;
    {
      const int m  = wave * 16 + l16;
      const int kk = half * 8;
      #pragma unroll
      for (int e2 = 0; e2 < 8; e2++) a[e2]     = bits_to_bf16(As[m][kk + e2]);
      #pragma unroll
      for (int e2 = 0; e2 < 8; e2++) a[8 + e2] = bits_to_bf16(As[m][16 + kk + e2]);
    }
    #pragma unroll
    for (int nt = 0; nt < NTILES; nt++) {
      bf16x16 b;
      const int nn = nt * 16 + l16;
      const int kk = half * 16;
      #pragma unroll
      for (int e2 = 0; e2 < 16; e2++) b[e2] = bits_to_bf16(Bs[nn][(e2 < 8) ? (half * 8 + e2) : (16 + half * 8 + (e2 - 8))]);
      acc[nt] = __builtin_amdgcn_wmma_f32_16x16x32_bf16(
          false, a, false, b, (short)0, acc[nt], false, false);
    }
    __syncthreads();
  }

  // D layout: element e -> row mRow0 + half*8 + e, col l16 (+16*nt)
  const long mRow0 = mBase + wave * 16;
  #pragma unroll
  for (int nt = 0; nt < NTILES; nt++) {
    const int nn = nt * 16 + l16;
    if (nn >= HH) continue;
    #pragma unroll
    for (int e2 = 0; e2 < 8; e2++) {
      long m = mRow0 + half * 8 + e2;
      if (m < nrows) {
        float v = acc[nt][e2];
        if (bias) v += bias[nn];
        if (relu) v = v > 0.0f ? v : 0.0f;
        Y[m * (long)ldy + yoff + nn] = v;
      }
    }
  }
}

// ---------------- atomic-free GCN aggregation: wave per destination node ----------------
// R[node, off + c] = bias[c] + sum_{incoming e} h[prow[e], c] * pnorm[e]
__global__ void __launch_bounds__(256) hgd_conv_csr(
    const float* __restrict__ h,
    const int* __restrict__ offs, const int* __restrict__ prow,
    const float* __restrict__ pnorm, const float* __restrict__ bias,
    float* __restrict__ R, int ldr, int off, int n)
{
  long node = (long)blockIdx.x * 8 + (threadIdx.x >> 5);
  if (node >= n) return;
  const int lane = threadIdx.x & 31;
  const int s0 = offs[node], s1 = offs[node + 1];
  const int c0 = lane, c1 = lane + 32, c2 = lane + 64, c3 = lane + 96;
  float a0 = bias[c0];
  float a1 = bias[c1];
  float a2 = bias[c2];
  float a3 = (c3 < HH) ? bias[c3] : 0.0f;
  for (int i = s0; i < s1; i++) {
    int r = prow[i];            // uniform across wave -> scalarized
    float nm = pnorm[i];
    const float* hr = h + (long)r * HH;
    a0 += hr[c0] * nm;          // coalesced 100-float row gather
    a1 += hr[c1] * nm;
    a2 += hr[c2] * nm;
    if (c3 < HH) a3 += hr[c3] * nm;
  }
  float* dst = R + node * (long)ldr + off;
  dst[c0] = a0;
  dst[c1] = a1;
  dst[c2] = a2;
  if (c3 < HH) dst[c3] = a3;
}

// ---------------- readout head: wave per node, accumulate into out ----------------
__global__ void __launch_bounds__(256) hgd_readout(
    const float* __restrict__ R, int ld,
    const float* __restrict__ Wr, const float* __restrict__ br,
    const float* __restrict__ wmix, float* __restrict__ out,
    int n, int init)
{
  long node = (long)blockIdx.x * 8 + (threadIdx.x >> 5);
  if (node >= n) return;
  int lane = threadIdx.x & 31;
  const float* rn = R + node * (long)ld;
  float s = 0.0f;
  for (int c = lane; c < ld; c += 32) s += rn[c] * Wr[c];
  #pragma unroll
  for (int o = 16; o > 0; o >>= 1) s += __shfl_down(s, o, 32);
  if (lane == 0) {
    float v = (s + br[0]) * wmix[0];
    if (init) out[node] = v;
    else      out[node] += v;
  }
}

extern "C" void kernel_launch(void* const* d_in, const int* in_sizes, int n_in,
                              void* d_out, int out_size, void* d_ws, size_t ws_size,
                              hipStream_t stream) {
  (void)n_in; (void)out_size; (void)ws_size;
  const float* x     = (const float*)d_in[0];
  const int*   ei1   = (const int*)d_in[1];
  const int*   ei2   = (const int*)d_in[2];
  const float* W1    = (const float*)d_in[3];  const float* b1    = (const float*)d_in[4];
  const float* Wk1_1 = (const float*)d_in[5];  const float* bk1_1 = (const float*)d_in[6];
  const float* Wk1_2 = (const float*)d_in[7];  const float* bk1_2 = (const float*)d_in[8];
  const float* Wk2_1 = (const float*)d_in[9];  const float* bk2_1 = (const float*)d_in[10];
  const float* Wk2_2 = (const float*)d_in[11]; const float* bk2_2 = (const float*)d_in[12];
  const float* Wk3_1 = (const float*)d_in[13]; const float* bk3_1 = (const float*)d_in[14];
  const float* Wk3_2 = (const float*)d_in[15]; const float* bk3_2 = (const float*)d_in[16];
  const float* Wr0 = (const float*)d_in[17];   const float* br0 = (const float*)d_in[18];
  const float* Wr1 = (const float*)d_in[19];   const float* br1 = (const float*)d_in[20];
  const float* Wr2 = (const float*)d_in[21];   const float* br2 = (const float*)d_in[22];
  const float* Wr3 = (const float*)d_in[23];   const float* br3 = (const float*)d_in[24];
  const float* w0  = (const float*)d_in[25];
  const float* w1  = (const float*)d_in[26];
  const float* w2  = (const float*)d_in[27];
  const float* w3  = (const float*)d_in[28];

  const int D_IN = 58;
  const long n  = in_sizes[0] / D_IN;
  const long E1 = in_sizes[1] / 2;
  const long E2 = in_sizes[2] / 2;

  // ---- workspace carve (256B aligned regions) ----
  char* base = (char*)d_ws;
  auto carve = [&](size_t bytes) -> char* {
    char* p = base;
    base += (bytes + 255) & ~(size_t)255;
    return p;
  };
  unsigned* cnt1 = (unsigned*)carve(n * 4);
  unsigned* cnt2 = (unsigned*)carve(n * 4);
  float* dis1 = (float*)carve(n * 4);
  float* dis2 = (float*)carve(n * 4);
  int* off1 = (int*)carve((n + 1) * 4);
  int* off2 = (int*)carve((n + 1) * 4);
  int* cur1 = (int*)carve(n * 4);
  int* cur2 = (int*)carve(n * 4);
  int* prow1 = (int*)carve(E1 * 4);
  int* prow2 = (int*)carve(E2 * 4);
  float* pnorm1 = (float*)carve(E1 * 4);
  float* pnorm2 = (float*)carve(E2 * 4);
  float* h1   = (float*)carve(n * HH * 4);
  float* h2   = (float*)carve(n * HH * 4);
  float* bufA = (float*)carve(n * 2 * HH * 4);
  float* bufB = (float*)carve(n * 2 * HH * 4);

  const int* row1 = ei1; const int* col1 = ei1 + E1;
  const int* row2 = ei2; const int* col2 = ei2 + E2;
  float* out = (float*)d_out;

  const int tpb = 256;
  const int gE1 = (int)((E1 + tpb - 1) / tpb);
  const int gE2 = (int)((E2 + tpb - 1) / tpb);
  const int gN  = (int)((n + tpb - 1) / tpb);
  const int gGemm = (int)((n + 127) / 128);
  const int gNode = (int)((n + 7) / 8);   // wave-per-node kernels

  // ---- CSR build (once; edge structure shared by all 3 layers) ----
  hipMemsetAsync(cnt1, 0, n * sizeof(unsigned), stream);
  hipMemsetAsync(cnt2, 0, n * sizeof(unsigned), stream);
  hgd_count_kernel<<<gE1, tpb, 0, stream>>>(col1, (int)E1, cnt1);
  hgd_count_kernel<<<gE2, tpb, 0, stream>>>(col2, (int)E2, cnt2);
  hgd_dis_kernel<<<gN, tpb, 0, stream>>>(cnt1, dis1, (int)n);
  hgd_dis_kernel<<<gN, tpb, 0, stream>>>(cnt2, dis2, (int)n);
  hgd_scan_kernel<<<1, 1024, 0, stream>>>(cnt1, off1, (int)n);
  hgd_scan_kernel<<<1, 1024, 0, stream>>>(cnt2, off2, (int)n);
  hgd_copy_int<<<gN, tpb, 0, stream>>>(cur1, off1, (int)n);
  hgd_copy_int<<<gN, tpb, 0, stream>>>(cur2, off2, (int)n);
  hgd_fill_kernel<<<gE1, tpb, 0, stream>>>(row1, col1, dis1, cur1, prow1, pnorm1, (int)E1);
  hgd_fill_kernel<<<gE2, tpb, 0, stream>>>(row2, col2, dis2, cur2, prow2, pnorm2, (int)E2);

  // ---- R0 = relu(x @ W1 + b1) -> bufA (ld 100) ----
  hgd_gemm_wmma<<<gGemm, tpb, 0, stream>>>(x, D_IN, D_IN, W1, bufA, HH, 0, (int)n, b1, 1);
  hgd_readout<<<gNode, tpb, 0, stream>>>(bufA, HH, Wr0, br0, w0, out, (int)n, 1);

  // ---- layer 1: R0(bufA, ld100) -> R1(bufB, ld200) ----
  hgd_gemm_wmma<<<gGemm, tpb, 0, stream>>>(bufA, HH, HH, Wk1_1, h1, HH, 0, (int)n, nullptr, 0);
  hgd_gemm_wmma<<<gGemm, tpb, 0, stream>>>(bufA, HH, HH, Wk1_2, h2, HH, 0, (int)n, nullptr, 0);
  hgd_conv_csr<<<gNode, tpb, 0, stream>>>(h1, off1, prow1, pnorm1, bk1_1, bufB, 2 * HH, 0,  (int)n);
  hgd_conv_csr<<<gNode, tpb, 0, stream>>>(h2, off2, prow2, pnorm2, bk1_2, bufB, 2 * HH, HH, (int)n);
  hgd_readout<<<gNode, tpb, 0, stream>>>(bufB, 2 * HH, Wr1, br1, w1, out, (int)n, 0);

  // ---- layer 2: R1(bufB) -> R2(bufA, ld200) ----
  hgd_gemm_wmma<<<gGemm, tpb, 0, stream>>>(bufB, 2 * HH, 2 * HH, Wk2_1, h1, HH, 0, (int)n, nullptr, 0);
  hgd_gemm_wmma<<<gGemm, tpb, 0, stream>>>(bufB, 2 * HH, 2 * HH, Wk2_2, h2, HH, 0, (int)n, nullptr, 0);
  hgd_conv_csr<<<gNode, tpb, 0, stream>>>(h1, off1, prow1, pnorm1, bk2_1, bufA, 2 * HH, 0,  (int)n);
  hgd_conv_csr<<<gNode, tpb, 0, stream>>>(h2, off2, prow2, pnorm2, bk2_2, bufA, 2 * HH, HH, (int)n);
  hgd_readout<<<gNode, tpb, 0, stream>>>(bufA, 2 * HH, Wr2, br2, w2, out, (int)n, 0);

  // ---- layer 3: R2(bufA) -> R3(bufB) ----
  hgd_gemm_wmma<<<gGemm, tpb, 0, stream>>>(bufA, 2 * HH, 2 * HH, Wk3_1, h1, HH, 0, (int)n, nullptr, 0);
  hgd_gemm_wmma<<<gGemm, tpb, 0, stream>>>(bufA, 2 * HH, 2 * HH, Wk3_2, h2, HH, 0, (int)n, nullptr, 0);
  hgd_conv_csr<<<gNode, tpb, 0, stream>>>(h1, off1, prow1, pnorm1, bk3_1, bufB, 2 * HH, 0,  (int)n);
  hgd_conv_csr<<<gNode, tpb, 0, stream>>>(h2, off2, prow2, pnorm2, bk3_2, bufB, 2 * HH, HH, (int)n);
  hgd_readout<<<gNode, tpb, 0, stream>>>(bufB, 2 * HH, Wr3, br3, w3, out, (int)n, 0);
}
